// SchemaGNN_15522011807978
// MI455X (gfx1250) — matrix operand order repaired and checked by reading, not compiled, not run on hardware
//
#include <hip/hip_runtime.h>
#include <hip/hip_bf16.h>
#include <math.h>

// ---------------------------------------------------------------------------
// MI455X / gfx1250 HGT GNN forward pass.
// Dense math: v_wmma_f32_16x16x32_bf16, B operands pre-swizzled into fragment
// order in global memory (two global_load_b128 per fragment, L2-resident),
// A operands staged fp32->bf16 through LDS with b128 loads/stores.
// Sparse (edge) math: L2-resident gathers + float atomics.
// ---------------------------------------------------------------------------

typedef __attribute__((ext_vector_type(16))) __bf16 v16bf;
typedef __attribute__((ext_vector_type(8)))  __bf16 v8bf;
typedef __attribute__((ext_vector_type(8)))  float  v8f;

#define HH 4
#define DDIM 64
#define CCH 256

__device__ __forceinline__ float gelu_tanh(float x) {
  return 0.5f * x * (1.f + tanhf(0.7978845608028654f * (x + 0.044715f * x * x * x)));
}

// Ordered float atomic max via signed/unsigned int atomics (buffer must be
// initialized to -inf = 0xFF800000).
__device__ __forceinline__ void atomicMaxF32(float* addr, float val) {
  if (val >= 0.f)
    atomicMax((int*)addr, __float_as_int(val));
  else
    atomicMin((unsigned int*)addr, (unsigned int)__float_as_int(val));
}

// ------------------------------ utility kernels ----------------------------

__global__ void k_fill(float* __restrict__ p, float v, size_t n) {
  size_t i = (size_t)blockIdx.x * blockDim.x + threadIdx.x;
  if (i < n) p[i] = v;
}

// Swizzle fp32 row-major W[K][N] (leading dim ldw) into bf16 WMMA-B fragment
// order: dst[((kt*NT + nt)*32 + lane)*16 + j] where the fragment mapping is
// lane -> col nt*16 + (lane&15), half j -> k = kt*32 + (j>>3)*16 + (lane>>4)*8 + (j&7).
// Each lane's 16 halves are then 32 contiguous bytes in global memory.
__global__ void k_swizzle_w(const float* __restrict__ src, __bf16* __restrict__ dst,
                            int K, int N, int ldw, long long sSrc, long long sDst) {
  src += (long long)blockIdx.z * sSrc;
  dst += (long long)blockIdx.z * sDst;
  const int NT = N >> 4;
  const int KT = (K + 31) >> 5;
  const int total = KT * NT * 512;
  int idx = blockIdx.x * blockDim.x + threadIdx.x;
  if (idx >= total) return;
  const int j = idx & 15;
  const int lane = (idx >> 4) & 31;
  const int t = idx >> 9;               // kt*NT + nt
  const int nt = t % NT, kt = t / NT;
  const int k = kt * 32 + (j >> 3) * 16 + (lane >> 4) * 8 + (j & 7);
  const int c = nt * 16 + (lane & 15);
  const float v = (k < K) ? src[(size_t)k * ldw + c] : 0.f;
  dst[idx] = (__bf16)v;
}

// ------------------------------ WMMA GEMM ----------------------------------
// Out[M,N] = act( A[M,K] (fp32, optional GELU on load) x Wsz (pre-swizzled
// bf16) + bias ).  Block tile: 64 rows x (32*NSUB) cols, 8 waves in a 4x2
// grid; each wave runs NSUB v_wmma_f32_16x16x32_bf16 per 32-wide k-step from
// one shared A fragment.  blockIdx.z batches via element strides sA/sW/sO.
// A-staging is branch-free: rows are clamped (rows >= M only feed accumulator
// rows that are never stored), K-tail uses clamped index + select.
template <int NSUB>
__global__ __launch_bounds__(256)
void k_gemm_wmma(const float* __restrict__ A, const __bf16* __restrict__ Wsz,
                 const float* __restrict__ bias, float* __restrict__ Out,
                 int M, int N, int K, int lda, int ldo,
                 long long sA, long long sW, long long sO,
                 int actA /*0 none, 2 gelu*/, int actOut /*0 none, 1 relu*/) {
  const int bz = blockIdx.z;
  A += (long long)bz * sA;
  Wsz += (long long)bz * sW;
  Out += (long long)bz * sO;

  // Row stride 40 halves = 80B (multiple of 16B) -> b128-friendly.
  __shared__ __bf16 As[64][40];

  const int tid = threadIdx.x;
  const int lane = tid & 31;
  const int wave = tid >> 5;
  const int wm = wave & 3;
  const int wn = wave >> 2;
  const int m0 = blockIdx.x * 64;
  const int n0 = blockIdx.y * (32 * NSUB);
  const int NT = N >> 4;

  v8f acc[NSUB];
#pragma unroll
  for (int s = 0; s < NSUB; ++s) acc[s] = (v8f){0.f, 0.f, 0.f, 0.f, 0.f, 0.f, 0.f, 0.f};

  const int ar = tid >> 2;            // A tile row 0..63
  const int ac = (tid & 3) * 8;       // A tile col base (8 elements)
  const int gr = m0 + ar;
  const int grc = gr < M ? gr : (M - 1);   // clamped row, always valid
  const float* arow = A + (size_t)grc * lda;

  const v16bf* wv = (const v16bf*)Wsz;
  const int ntBase = (n0 >> 4) + wn * NSUB;

  for (int k0 = 0; k0 < K; k0 += 32) {
    // ---- stage A tile (64x32 fp32 -> bf16); branch-free boundary handling
    {
      float f[8];
      const int gk0 = k0 + ac;
      if (gk0 + 7 < K) {
        const float4 p0 = *(const float4*)(arow + gk0);
        const float4 p1 = *(const float4*)(arow + gk0 + 4);
        f[0] = p0.x; f[1] = p0.y; f[2] = p0.z; f[3] = p0.w;
        f[4] = p1.x; f[5] = p1.y; f[6] = p1.z; f[7] = p1.w;
      } else {
#pragma unroll
        for (int j = 0; j < 8; ++j) {
          const int gk = gk0 + j;
          const int gkc = gk < K ? gk : (K - 1);
          const float v = arow[gkc];            // unconditional, in-bounds
          f[j] = (gk < K) ? v : 0.f;            // v_cndmask
        }
      }
      v8bf st;
      if (actA == 2) {
#pragma unroll
        for (int j = 0; j < 8; ++j) st[j] = (__bf16)gelu_tanh(f[j]);
      } else {
#pragma unroll
        for (int j = 0; j < 8; ++j) st[j] = (__bf16)f[j];
      }
      *(v8bf*)&As[ar][ac] = st;
    }
    __syncthreads();

    // ---- A fragment (ISA 7.12.2, 16-bit A 16x32): lane l -> row wm*16+(l&15)
    // halves 0..7 = K[(l>>4)*8 .. +7], halves 8..15 = K[16+(l>>4)*8 .. +7]
    const __bf16* ap = &As[wm * 16 + (lane & 15)][(lane >> 4) * 8];
    const v8bf a0 = *(const v8bf*)ap;         // ds_load_b128
    const v8bf a1 = *(const v8bf*)(ap + 16);  // ds_load_b128
    const v16bf afrag = __builtin_shufflevector(a0, a1, 0, 1, 2, 3, 4, 5, 6, 7,
                                                8, 9, 10, 11, 12, 13, 14, 15);

    // ---- B fragments straight from pre-swizzled global (2x b128 each)
    const long long wb = ((long long)(k0 >> 5) * NT + ntBase) * 32 + lane;
#pragma unroll
    for (int s = 0; s < NSUB; ++s) {
      const v16bf bfrag = wv[wb + (long long)s * 32];
      acc[s] = __builtin_amdgcn_wmma_f32_16x16x32_bf16(false, afrag, false, bfrag,
                                                       (short)0, acc[s], false, false);
    }
    __syncthreads();
  }

  // ---- store: C/D layout VGPR i -> row i + 8*(lane>>4), col lane&15
  const int rbase = m0 + wm * 16 + (lane >> 4) * 8;
  const int cb = n0 + wn * 16 * NSUB + (lane & 15);
#pragma unroll
  for (int s = 0; s < NSUB; ++s) {
    const int c = cb + s * 16;
    if (c >= N) continue;
    const float bv = bias ? bias[c] : 0.f;
#pragma unroll
    for (int i = 0; i < 8; ++i) {
      const int r = rbase + i;
      if (r >= M) continue;
      float v = acc[s][i] + bv;
      if (actOut == 1) v = fmaxf(v, 0.f);
      Out[(size_t)r * ldo + c] = v;
    }
  }
}

// ------------------------------ edge kernels -------------------------------

// att[e,h] = dot64(kA[src[e],h,:], q[dst[e],h,:]) * p_rel[h] / sqrt(D);
// also segment-max into m[dst,h].
__global__ void k_edge_att(const float* __restrict__ kA, const float* __restrict__ kqv,
                           const int* __restrict__ src, const int* __restrict__ dst,
                           const float* __restrict__ prel, float* __restrict__ att,
                           float* __restrict__ m, int E) {
  int idx = blockIdx.x * blockDim.x + threadIdx.x;
  if (idx >= E * HH) return;
  const int e = idx >> 2, hh = idx & 3;
  const int s = src[e], d = dst[e];
  const float4* kp = (const float4*)(kA + (size_t)s * CCH + hh * DDIM);
  const float4* qp = (const float4*)(kqv + (size_t)d * 768 + 256 + hh * DDIM);
  float acc = 0.f;
#pragma unroll
  for (int j = 0; j < DDIM / 4; ++j) {
    const float4 a = kp[j], b = qp[j];
    acc += a.x * b.x + a.y * b.y + a.z * b.z + a.w * b.w;
  }
  acc *= prel[hh] * 0.125f;  // 1/sqrt(64)
  att[idx] = acc;
  atomicMaxF32(&m[(size_t)d * HH + hh], acc);
}

// att[e,h] = exp(att - m[dst]); z[dst,h] += att
__global__ void k_edge_exp(float* __restrict__ att, const int* __restrict__ dst,
                           const float* __restrict__ m, float* __restrict__ z, int E) {
  int idx = blockIdx.x * blockDim.x + threadIdx.x;
  if (idx >= E * HH) return;
  const int e = idx >> 2, hh = idx & 3;
  const int d = dst[e];
  const float ee = expf(att[idx] - m[(size_t)d * HH + hh]);
  att[idx] = ee;
  atomicAdd(&z[(size_t)d * HH + hh], ee);
}

// out[dst, c] += (att[e,h]/(z[dst,h]+eps)) * vM[src, c]
// 4 edges per block; each thread owns 4 channels (float4 gather, 4 atomics).
__global__ void k_edge_scatter(const float* __restrict__ vM, const int* __restrict__ src,
                               const int* __restrict__ dst, const float* __restrict__ att,
                               const float* __restrict__ z, float* __restrict__ out, int E) {
  const int e = blockIdx.x * 4 + (threadIdx.x >> 6);
  if (e >= E) return;
  const int c = (threadIdx.x & 63) * 4;
  const int hh = c >> 6;
  const int s = src[e], d = dst[e];
  const float alpha = att[(size_t)e * HH + hh] / (z[(size_t)d * HH + hh] + 1e-16f);
  const float4 v = *(const float4*)(vM + (size_t)s * CCH + c);
  float* o = out + (size_t)d * CCH + c;
  atomicAdd(o + 0, alpha * v.x);
  atomicAdd(o + 1, alpha * v.y);
  atomicAdd(o + 2, alpha * v.z);
  atomicAdd(o + 3, alpha * v.w);
}

// h = LayerNorm(h + a*proj + (1-a)*h) with a = sigmoid(skip)
__global__ void k_epilogue_ln(float* __restrict__ h, const float* __restrict__ proj,
                              const float* __restrict__ skipp, const float* __restrict__ g,
                              const float* __restrict__ bta) {
  const int n = blockIdx.x, c = threadIdx.x;
  const float a = 1.f / (1.f + expf(-skipp[0]));
  const size_t idx = (size_t)n * CCH + c;
  const float x = h[idx], p = proj[idx];
  const float t = x + a * p + (1.f - a) * x;
  __shared__ float s1[CCH];
  __shared__ float s2[CCH];
  s1[c] = t;
  s2[c] = t * t;
  __syncthreads();
  for (int off = 128; off > 0; off >>= 1) {
    if (c < off) { s1[c] += s1[c + off]; s2[c] += s2[c + off]; }
    __syncthreads();
  }
  const float mean = s1[0] * (1.f / CCH);
  const float var = s2[0] * (1.f / CCH) - mean * mean;
  h[idx] = (t - mean) * rsqrtf(var + 1e-5f) * g[c] + bta[c];
}

// ------------------------------ pooling / heads ----------------------------

__global__ void k_pool_accum(const float* __restrict__ h, const int* __restrict__ batch,
                             float* __restrict__ gsum, float* __restrict__ gmax, int N) {
  const int n = blockIdx.x;
  if (n >= N) return;
  const int c = threadIdx.x;
  const float v = h[(size_t)n * CCH + c];
  const int g = batch[n];
  atomicAdd(&gsum[(size_t)g * CCH + c], v);
  atomicMaxF32(&gmax[(size_t)g * CCH + c], v);
}

__global__ void k_pool_count(const int* __restrict__ batch, float* __restrict__ counts, int N) {
  int i = blockIdx.x * blockDim.x + threadIdx.x;
  if (i < N) atomicAdd(&counts[batch[i]], 1.f);
}

__global__ void k_build_ge(const float* __restrict__ gsum, const float* __restrict__ gmax,
                           const float* __restrict__ counts, float* __restrict__ ge, int B) {
  int idx = blockIdx.x * blockDim.x + threadIdx.x;
  if (idx >= B * 2 * CCH) return;
  const int b = idx >> 9, j = idx & 511;
  const float cnt = counts[b];
  float v;
  if (j < CCH) v = gsum[(size_t)b * CCH + j] / fmaxf(cnt, 1.f);
  else         v = (cnt > 0.f) ? gmax[(size_t)b * CCH + (j - CCH)] : 0.f;
  ge[idx] = v;
}

// out_raw[i] = A[i,:] . w + b ; out_sig = sigmoid(out_raw)
__global__ void k_dot_head(const float* __restrict__ A, const float* __restrict__ wv,
                           const float* __restrict__ bsc, float* __restrict__ out_sig,
                           float* __restrict__ out_raw, int M, int K) {
  int i = blockIdx.x * blockDim.x + threadIdx.x;
  if (i >= M) return;
  const float4* a = (const float4*)(A + (size_t)i * K);
  const float4* w = (const float4*)wv;
  float s = bsc[0];
  for (int j = 0; j < K / 4; ++j) {
    const float4 av = a[j], wvv = w[j];
    s += av.x * wvv.x + av.y * wvv.y + av.z * wvv.z + av.w * wvv.w;
  }
  out_raw[i] = s;
  out_sig[i] = 1.f / (1.f + expf(-s));
}

// ------------------------------ host driver --------------------------------

extern "C" void kernel_launch(void* const* d_in, const int* in_sizes, int n_in,
                              void* d_out, int out_size, void* d_ws, size_t ws_size,
                              hipStream_t stream) {
  const int N = in_sizes[0] / 404;   // 100000
  const int E = in_sizes[1] / 2;     // 200000
  const int B = 512;
  const int R = 5, L = 3;

  const float* x = (const float*)d_in[0];
  const int* ei[5];
  for (int r = 0; r < R; ++r) ei[r] = (const int*)d_in[1 + r];
  const int* batch = (const int*)d_in[6];

  // params leaves in pytree sorted-key order:
  // g1{b,w}, g2{b,w}, in_proj{b,w}, l1{b,w}, l2{b,w},
  // layers[i]{a_rel, kqv.b, kqv.w, ln_b, ln_g, m_rel, out.b, out.w, p_rel, skip}
  const float* g1b = (const float*)d_in[7];
  const float* g1w = (const float*)d_in[8];
  const float* g2b = (const float*)d_in[9];
  const float* g2w = (const float*)d_in[10];
  const float* inb = (const float*)d_in[11];
  const float* inw = (const float*)d_in[12];
  const float* l1b = (const float*)d_in[13];
  const float* l1w = (const float*)d_in[14];
  const float* l2b = (const float*)d_in[15];
  const float* l2w = (const float*)d_in[16];
  const float *arel[3], *kqvb[3], *kqvw[3], *lnb[3], *lng[3], *mrel[3],
              *outb[3], *outw[3], *prel[3], *skip[3];
  for (int l = 0; l < L; ++l) {
    const int base = 17 + l * 10;
    arel[l] = (const float*)d_in[base + 0];
    kqvb[l] = (const float*)d_in[base + 1];
    kqvw[l] = (const float*)d_in[base + 2];
    lnb[l]  = (const float*)d_in[base + 3];
    lng[l]  = (const float*)d_in[base + 4];
    mrel[l] = (const float*)d_in[base + 5];
    outb[l] = (const float*)d_in[base + 6];
    outw[l] = (const float*)d_in[base + 7];
    prel[l] = (const float*)d_in[base + 8];
    skip[l] = (const float*)d_in[base + 9];
  }

  // ---- workspace carve-out (all offsets 256B aligned) ----
  char* wp = (char*)d_ws;
  auto alloc = [&](size_t bytes) -> void* {
    void* p = (void*)wp;
    wp += (bytes + 255) & ~(size_t)255;
    return p;
  };
  float* hbuf   = (float*)alloc((size_t)N * 256 * 4);
  float* kqvbuf = (float*)alloc((size_t)N * 768 * 4);
  float* tbuf   = (float*)alloc((size_t)N * 256 * 4);   // kA / vM / proj (reused)
  float* attn   = (float*)alloc((size_t)N * 256 * 4);
  float* attbuf = (float*)alloc((size_t)R * E * HH * 4);
  float* mbuf   = (float*)alloc((size_t)N * HH * 4);
  float* zbuf   = (float*)alloc((size_t)N * HH * 4);
  float* gsum   = (float*)alloc((size_t)B * 256 * 4);
  float* gmax   = (float*)alloc((size_t)B * 256 * 4);
  float* counts = (float*)alloc((size_t)B * 4);
  float* gebuf  = (float*)alloc((size_t)B * 512 * 4);
  float* ge2    = (float*)alloc((size_t)B * 256 * 4);
  // swizzled bf16 weights: KT*NT*512 elements each
  __bf16* inw_sz = (__bf16*)alloc((size_t)13 * 16 * 512 * 2);   // K=404,N=256
  __bf16* g1w_sz = (__bf16*)alloc((size_t)16 * 16 * 512 * 2);   // K=512,N=256
  __bf16* l1w_sz = (__bf16*)alloc((size_t)8 * 8 * 512 * 2);     // K=256,N=128
  __bf16 *kqvw_sz[3], *outw_sz[3], *arel_sz[3], *mrel_sz[3];
  for (int l = 0; l < L; ++l) {
    kqvw_sz[l] = (__bf16*)alloc((size_t)8 * 48 * 512 * 2);      // K=256,N=768
    outw_sz[l] = (__bf16*)alloc((size_t)8 * 16 * 512 * 2);      // K=256,N=256
    arel_sz[l] = (__bf16*)alloc((size_t)R * HH * 4096 * 2);     // 20x (K=64,N=64)
    mrel_sz[l] = (__bf16*)alloc((size_t)R * HH * 4096 * 2);
  }
  float* l1buf = kqvbuf;  // reuse: kqv dead after the layer stack

  auto cdiv = [](long long a, long long b) { return (int)((a + b - 1) / b); };
  auto swz = [&](const float* s, __bf16* d, int K, int Nc, int nz, long long ss, long long sd) {
    const int total = ((K + 31) / 32) * (Nc / 16) * 512;
    k_swizzle_w<<<dim3(cdiv(total, 256), 1, nz), 256, 0, stream>>>(s, d, K, Nc, Nc, ss, sd);
  };

  // ---- weight swizzle fp32 -> bf16 fragment order ----
  swz(inw, inw_sz, 404, 256, 1, 0, 0);
  swz(g1w, g1w_sz, 512, 256, 1, 0, 0);
  swz(l1w, l1w_sz, 256, 128, 1, 0, 0);
  for (int l = 0; l < L; ++l) {
    swz(kqvw[l], kqvw_sz[l], 256, 768, 1, 0, 0);
    swz(outw[l], outw_sz[l], 256, 256, 1, 0, 0);
    swz(arel[l], arel_sz[l], 64, 64, R * HH, 4096, 4096);   // per (r,h) 64x64 block
    swz(mrel[l], mrel_sz[l], 64, 64, R * HH, 4096, 4096);
  }

  const int mt = cdiv(N, 64);  // row tiles for node-dim GEMMs

  // ---- h = relu(x @ in_proj.w + b) ----
  k_gemm_wmma<4><<<dim3(mt, 2, 1), 256, 0, stream>>>(
      x, inw_sz, inb, hbuf, N, 256, 404, 404, 256, 0, 0, 0, 0, 1);

  // ---- layer stack ----
  for (int l = 0; l < L; ++l) {
    // kqv = h @ kqv.w + b   -> [N,768]  (k|q|v)
    k_gemm_wmma<4><<<dim3(mt, 6, 1), 256, 0, stream>>>(
        hbuf, kqvw_sz[l], kqvb[l], kqvbuf, N, 768, 256, 256, 768, 0, 0, 0, 0, 0);

    k_fill<<<cdiv((long long)N * HH, 256), 256, 0, stream>>>(
        mbuf, -__builtin_huge_valf(), (size_t)N * HH);
    hipMemsetAsync(zbuf, 0, (size_t)N * HH * 4, stream);

    // attention logits + segment max, per relation
    for (int r = 0; r < R; ++r) {
      // kA = k @ a_rel[r]  (batched over heads via blockIdx.z)
      k_gemm_wmma<2><<<dim3(mt, 1, HH), 256, 0, stream>>>(
          kqvbuf /*k slice*/, arel_sz[l] + (size_t)r * HH * 4096, nullptr, tbuf,
          N, 64, 64, 768, 256, /*sA=*/64, /*sW=*/4096, /*sO=*/64, 0, 0);
      k_edge_att<<<cdiv((long long)E * HH, 256), 256, 0, stream>>>(
          tbuf, kqvbuf, ei[r], ei[r] + E, prel[l] + r * HH,
          attbuf + (size_t)r * E * HH, mbuf, E);
    }
    // segment softmax numerator + denominator
    for (int r = 0; r < R; ++r)
      k_edge_exp<<<cdiv((long long)E * HH, 256), 256, 0, stream>>>(
          attbuf + (size_t)r * E * HH, ei[r] + E, mbuf, zbuf, E);

    hipMemsetAsync(attn, 0, (size_t)N * 256 * 4, stream);
    // message scatter, per relation
    for (int r = 0; r < R; ++r) {
      // vM = v @ m_rel[r]
      k_gemm_wmma<2><<<dim3(mt, 1, HH), 256, 0, stream>>>(
          kqvbuf + 512 /*v slice*/, mrel_sz[l] + (size_t)r * HH * 4096, nullptr, tbuf,
          N, 64, 64, 768, 256, 64, 4096, 64, 0, 0);
      k_edge_scatter<<<cdiv(E, 4), 256, 0, stream>>>(
          tbuf, ei[r], ei[r] + E, attbuf + (size_t)r * E * HH, zbuf, attn, E);
    }

    // proj = gelu(attn) @ out.w + b  (GELU fused into A-load)
    k_gemm_wmma<4><<<dim3(mt, 2, 1), 256, 0, stream>>>(
        attn, outw_sz[l], outb[l], tbuf, N, 256, 256, 256, 256, 0, 0, 0, 2, 0);

    // h = LN(h + a*proj + (1-a)*h)
    k_epilogue_ln<<<N, 256, 0, stream>>>(hbuf, tbuf, skip[l], lng[l], lnb[l]);
  }

  // ---- pooling ----
  hipMemsetAsync(gsum, 0, (size_t)B * 256 * 4, stream);
  hipMemsetAsync(counts, 0, (size_t)B * 4, stream);
  k_fill<<<cdiv((long long)B * 256, 256), 256, 0, stream>>>(
      gmax, -__builtin_huge_valf(), (size_t)B * 256);
  k_pool_accum<<<N, 256, 0, stream>>>(hbuf, batch, gsum, gmax, N);
  k_pool_count<<<cdiv(N, 256), 256, 0, stream>>>(batch, counts, N);
  k_build_ge<<<cdiv((long long)B * 512, 256), 256, 0, stream>>>(gsum, gmax, counts, gebuf, B);

  // ---- heads ----
  float* out = (float*)d_out;  // [sig(vl) B | vl B | sig(nl) N | nl N]
  k_gemm_wmma<4><<<dim3(cdiv(B, 64), 2, 1), 256, 0, stream>>>(
      gebuf, g1w_sz, g1b, ge2, B, 256, 512, 512, 256, 0, 0, 0, 0, 1);
  k_dot_head<<<cdiv(B, 256), 256, 0, stream>>>(ge2, g2w, g2b, out, out + B, B, 256);
  k_gemm_wmma<4><<<dim3(mt, 1, 1), 256, 0, stream>>>(
      hbuf, l1w_sz, l1b, l1buf, N, 128, 256, 256, 128, 0, 0, 0, 0, 1);
  k_dot_head<<<cdiv(N, 256), 256, 0, stream>>>(l1buf, l2w, l2b,
                                               out + 2 * B, out + 2 * B + N, N, 128);
}
